// TransformerBlock_36876589203654
// MI455X (gfx1250) — matrix operand order repaired and checked by reading, not compiled, not run on hardware
//
#include <hip/hip_runtime.h>

typedef __attribute__((ext_vector_type(16))) _Float16 v16h;
typedef __attribute__((ext_vector_type(8)))  _Float16 v8h;
typedef __attribute__((ext_vector_type(8)))  float    v8f;

#define WAVE_LDS_FENCE() asm volatile("s_wait_dscnt 0" ::: "memory")

namespace {
constexpr int BQ   = 2;
constexpr int NPT  = 16384;
constexpr int ROWS = BQ * NPT;          // 32768
constexpr int KNN  = 24;
constexpr int DPTS = 64;
constexpr int DM   = 128;

// f16 LDS tile strides (bank-conflict avoiding)
constexpr int LDA   = 136;              // halfs; 272B/row -> rows spread 4 banks apart
constexpr int LDATT = 132;              // floats

// per-wave LDS scratch in fused kernel
constexpr int HBUF_HALF   = 32 * LDA;                 // 4352 halfs = 8704 B
constexpr int ATT_FLOATS  = KNN * LDATT;              // 3168 floats = 12672 B
constexpr int PER_WAVE_B  = 3 * HBUF_HALF * 2 + ATT_FLOATS * 4;   // 38784 B
constexpr int FUSED_WAVES = 4;
constexpr int SMEM_FUSED  = FUSED_WAVES * PER_WAVE_B;             // 155136 B

// workspace layout (bytes)
constexpr size_t WS_X      = 0;                                   // ROWS*DM f32 (16 MB)
constexpr size_t WS_WT_FC1 = (size_t)ROWS * DM * 4;
constexpr size_t WS_WT_D1  = WS_WT_FC1 + 128 * 64 * 2;
constexpr size_t WS_WT_D2  = WS_WT_D1  + 128 * 64 * 2;
constexpr size_t WS_WT_G1  = WS_WT_D2  + 128 * 128 * 2;
constexpr size_t WS_WT_G2  = WS_WT_G1  + 128 * 128 * 2;
constexpr size_t WS_WT_FC2 = WS_WT_G2  + 128 * 128 * 2;
} // namespace

// ---------- WMMA fragment helpers (CDNA5 wave32 layouts, ISA 7.12.2) ----------

// A fragment 16x32 f16 from a row-major f16 buffer (LDS): lane<16 holds row=lane,
// k = kh..kh+15 contiguous, kh = 16*(lane>=16). Two contiguous 16B chunks.
__device__ __forceinline__ v16h ldsA(const _Float16* buf, int ld, int mt, int ks, int lane) {
    int row = mt * 16 + (lane & 15);
    int kh  = (lane >> 4) * 16;
    const _Float16* p = buf + row * ld + ks * 32 + kh;
    v8h lo = *(const v8h*)(p);
    v8h hi = *(const v8h*)(p + 8);
    v16h a;
#pragma unroll
    for (int i = 0; i < 8; ++i) { a[i] = lo[i]; a[i + 8] = hi[i]; }
    return a;
}

// B fragment 32x16 f16 from a TRANSPOSED (NxK row-major) f16 weight matrix:
// lane holds column n = nt*16 + (lane&15), k = kh..kh+15 contiguous, kh = 16*(lane>=16).
__device__ __forceinline__ v16h glbB(const _Float16* wt, int ldk, int nt, int ks, int lane) {
    int col = nt * 16 + (lane & 15);
    int kh  = (lane >> 4) * 16;
    const _Float16* p = wt + (size_t)col * ldk + ks * 32 + kh;
    v8h lo = *(const v8h*)(p);
    v8h hi = *(const v8h*)(p + 8);
    v16h b;
#pragma unroll
    for (int i = 0; i < 8; ++i) { b[i] = lo[i]; b[i + 8] = hi[i]; }
    return b;
}

__device__ __forceinline__ v8f wmma_f16(v16h a, v16h b, v8f c) {
    return __builtin_amdgcn_wmma_f32_16x16x32_f16(false, a, false, b, (short)0, c, false, false);
}

// ---------- fused per-point GEMM stages (M=24 pad 32, N=128) ----------

template <int KSTEPS, bool RELU>
__device__ __forceinline__ void gemm_lds(const _Float16* Ab, const _Float16* Wt, int ldk,
                                         const float* __restrict__ bias, _Float16* Ob, int lane) {
    for (int nt = 0; nt < 8; ++nt) {
        float bv = bias[nt * 16 + (lane & 15)];
        v8f acc0 = {};
        v8f acc1 = {};
#pragma unroll
        for (int ks = 0; ks < KSTEPS; ++ks) {
            v16h bf = glbB(Wt, ldk, nt, ks, lane);
            v16h a0 = ldsA(Ab, LDA, 0, ks, lane);
            v16h a1 = ldsA(Ab, LDA, 1, ks, lane);
            acc0 = wmma_f16(a0, bf, acc0);
            acc1 = wmma_f16(a1, bf, acc1);
        }
        // D layout: vgpr r -> M = rb+r, lane -> N = col (C/D 16x16 table)
        int col = nt * 16 + (lane & 15);
        int rb  = (lane >> 4) * 8;
#pragma unroll
        for (int r = 0; r < 8; ++r) {
            float v0 = acc0[r] + bv;
            float v1 = acc1[r] + bv;
            if (RELU) { v0 = fmaxf(v0, 0.f); v1 = fmaxf(v1, 0.f); }
            Ob[(rb + r) * LDA + col]        = (_Float16)v0;
            Ob[(16 + rb + r) * LDA + col]   = (_Float16)v1;
        }
    }
}

__device__ __forceinline__ void gemm_attn(const _Float16* Ab, const _Float16* Wt,
                                          const float* __restrict__ bias, float* At, int lane) {
    for (int nt = 0; nt < 8; ++nt) {
        float bv = bias[nt * 16 + (lane & 15)];
        v8f acc0 = {};
        v8f acc1 = {};
#pragma unroll
        for (int ks = 0; ks < 4; ++ks) {
            v16h bf = glbB(Wt, 128, nt, ks, lane);
            v16h a0 = ldsA(Ab, LDA, 0, ks, lane);
            v16h a1 = ldsA(Ab, LDA, 1, ks, lane);
            acc0 = wmma_f16(a0, bf, acc0);
            acc1 = wmma_f16(a1, bf, acc1);
        }
        int col = nt * 16 + (lane & 15);
        int rb  = (lane >> 4) * 8;
#pragma unroll
        for (int r = 0; r < 8; ++r) {
            int r0 = rb + r;
            if (r0 < KNN) At[r0 * LDATT + col] = acc0[r] + bv;
            int r1 = 16 + rb + r;
            if (r1 < KNN) At[r1 * LDATT + col] = acc1[r] + bv;
        }
    }
}

// ---------- kernel 0: weight conversion to transposed f16 ----------

__global__ void prep_weights(const float* __restrict__ fc1_w, const float* __restrict__ d1_w,
                             const float* __restrict__ d2_w, const float* __restrict__ g1_w,
                             const float* __restrict__ g2_w, const float* __restrict__ fc2_w,
                             char* __restrict__ ws) {
    int i = blockIdx.x * blockDim.x + threadIdx.x;
    _Float16* wfc1 = (_Float16*)(ws + WS_WT_FC1);
    _Float16* wd1  = (_Float16*)(ws + WS_WT_D1);
    _Float16* wd2  = (_Float16*)(ws + WS_WT_D2);
    _Float16* wg1  = (_Float16*)(ws + WS_WT_G1);
    _Float16* wg2  = (_Float16*)(ws + WS_WT_G2);
    _Float16* wfc2 = (_Float16*)(ws + WS_WT_FC2);
    if (i < 8192) {                       // fc1: (64,128) -> T (128,64)
        int n = i >> 6, k = i & 63;
        wfc1[i] = (_Float16)fc1_w[k * DM + n];
    } else if (i < 16384) {               // d1: (60,128) -> T (128,64) zero-padded K
        int j = i - 8192;
        int n = j >> 6, k = j & 63;
        wd1[j] = (k < 60) ? (_Float16)d1_w[k * DM + n] : (_Float16)0.f;
    } else if (i < 16384 + 4 * 16384) {   // d2,g1,g2,fc2: (128,128) -> T
        int j = i - 16384;
        int w = j >> 14;
        int t = j & 16383;
        int n = t >> 7, k = t & 127;
        const float* src = (w == 0) ? d2_w : (w == 1) ? g1_w : (w == 2) ? g2_w : fc2_w;
        _Float16*    dst = (w == 0) ? wd2  : (w == 1) ? wg1  : (w == 2) ? wg2  : wfc2;
        dst[t] = (_Float16)src[k * DM + n];
    }
}

// ---------- kernel 1: x = features @ fc1 + b ----------

__global__ void __launch_bounds__(256) fc1_kernel(const float* __restrict__ features,
                                                  const _Float16* __restrict__ wt,
                                                  const float* __restrict__ bias,
                                                  float* __restrict__ xout) {
    __shared__ __align__(16) _Float16 As[8][16 * 72];
    int lane = threadIdx.x & 31, wave = threadIdx.x >> 5;
    int mbase = blockIdx.x * 128 + wave * 16;
    _Float16* aw = As[wave];
    for (int t = lane; t < 16 * DPTS; t += 32) {
        int r = t >> 6, c = t & 63;
        aw[r * 72 + c] = (_Float16)features[(size_t)(mbase + r) * DPTS + c];
    }
    WAVE_LDS_FENCE();
    for (int nt = 0; nt < 8; ++nt) {
        float bv = bias[nt * 16 + (lane & 15)];
        v8f acc = {};
#pragma unroll
        for (int ks = 0; ks < 2; ++ks) {
            v16h bf = glbB(wt, 64, nt, ks, lane);
            v16h a  = ldsA(aw, 72, 0, ks, lane);
            acc = wmma_f16(a, bf, acc);
        }
        int col = nt * 16 + (lane & 15);
        int rb  = (lane >> 4) * 8;
#pragma unroll
        for (int r = 0; r < 8; ++r)
            xout[(size_t)(mbase + rb + r) * DM + col] = acc[r] + bv;
    }
}

// ---------- kernel 2: fused per-point transformer core (1 wave = 1 point) ----------

__global__ void __launch_bounds__(128) pt_fused(
    const float* __restrict__ xyz, const int* __restrict__ knn_idx,
    const float* __restrict__ knn_xyz, const float* __restrict__ xf,
    const _Float16* __restrict__ wd1, const float* __restrict__ bd1,
    const _Float16* __restrict__ wd2, const float* __restrict__ bd2,
    const _Float16* __restrict__ wg1, const float* __restrict__ bg1,
    const _Float16* __restrict__ wg2, const float* __restrict__ bg2,
    float* __restrict__ attn_out, float* __restrict__ res_out) {
    extern __shared__ __align__(16) char smem[];
    int lane = threadIdx.x & 31, wave = threadIdx.x >> 5;
    char* base    = smem + (size_t)wave * PER_WAVE_B;
    _Float16* Ab  = (_Float16*)base;                 // emb -> h
    _Float16* Tb  = Ab + HBUF_HALF;                  // t1 -> t2
    _Float16* Pb  = Tb + HBUF_HALF;                  // pos_enc -> kv+pos_enc
    float*    At  = (float*)(base + 3 * HBUF_HALF * 2); // attn logits

    int p = blockIdx.x * FUSED_WAVES + wave;         // global point id, < 32768
    int b = p >> 14;                                 // batch

    // --- sincos positional embedding: rows 0..23, cols 0..59 of Ab ---
    float cx = xyz[(size_t)p * 3 + 0];
    float cy = xyz[(size_t)p * 3 + 1];
    float cz = xyz[(size_t)p * 3 + 2];
    if (lane < 30) {
        int d = lane / 10, j = lane - d * 10;
        float omega = __expf(-(float)j * 0.92103403719761836f); // 10000^(-j/10)
        float cc = (d == 0) ? cx : (d == 1) ? cy : cz;
        for (int k = 0; k < KNN; ++k) {
            float g = cc - knn_xyz[((size_t)p * KNN + k) * 3 + d];
            float t = g * omega;
            Ab[k * LDA + d * 20 + j]      = (_Float16)__sinf(t);
            Ab[k * LDA + d * 20 + 10 + j] = (_Float16)__cosf(t);
        }
    }
    WAVE_LDS_FENCE();

    gemm_lds<2, true>(Ab, wd1, 64, bd1, Tb, lane);   // t1 = relu(emb @ d1 + b)
    WAVE_LDS_FENCE();
    gemm_lds<4, false>(Tb, wd2, 128, bd2, Pb, lane); // pos_enc = t1 @ d2 + b
    WAVE_LDS_FENCE();

    // --- gather kv, build h = x - kv + pos, kvpos = kv + pos ---
    const float* xrow = xf + (size_t)p * DM;
    float xm[4];
#pragma unroll
    for (int i = 0; i < 4; ++i) xm[i] = xrow[lane + 32 * i];
    for (int k = 0; k < KNN; ++k) {
        int idx = knn_idx[(size_t)p * KNN + k];
        const float* kvr = xf + ((size_t)b * NPT + idx) * DM;
#pragma unroll
        for (int i = 0; i < 4; ++i) {
            int c = lane + 32 * i;
            float kvv = kvr[c];
            float pe  = (float)Pb[k * LDA + c];
            Ab[k * LDA + c] = (_Float16)(xm[i] - kvv + pe);
            Pb[k * LDA + c] = (_Float16)(kvv + pe);
        }
    }
    WAVE_LDS_FENCE();

    gemm_lds<4, true>(Ab, wg1, 128, bg1, Tb, lane);  // t2 = relu(h @ g1 + b)
    WAVE_LDS_FENCE();
    gemm_attn(Tb, wg2, bg2, At, lane);               // logits = t2 @ g2 + b
    WAVE_LDS_FENCE();

    // --- softmax over K per channel + weighted sum ---
    const float scale = 0.08838834764831845f;        // 1/sqrt(128)
#pragma unroll
    for (int i = 0; i < 4; ++i) {
        int f = lane + 32 * i;
        float mx = -3.4e38f;
        for (int k = 0; k < KNN; ++k) mx = fmaxf(mx, At[k * LDATT + f] * scale);
        float s = 0.f;
        for (int k = 0; k < KNN; ++k) s += __expf(At[k * LDATT + f] * scale - mx);
        float rs  = 1.f / s;
        float acc = 0.f;
        for (int k = 0; k < KNN; ++k) {
            float a = __expf(At[k * LDATT + f] * scale - mx) * rs;
            attn_out[((size_t)p * KNN + k) * DM + f] = a;
            acc += a * (float)Pb[k * LDA + f];
        }
        res_out[(size_t)p * DM + f] = acc;           // pre-fc2 partial result
    }
}

// ---------- kernel 3: res = res_pre @ fc2 + b + x (in-place over res region) ----------

__global__ void __launch_bounds__(256) fc2_kernel(const float* __restrict__ resp,
                                                  const _Float16* __restrict__ wt,
                                                  const float* __restrict__ bias,
                                                  const float* __restrict__ xres,
                                                  float* __restrict__ out) {
    __shared__ __align__(16) _Float16 As[8][16 * LDA];
    int lane = threadIdx.x & 31, wave = threadIdx.x >> 5;
    int mbase = blockIdx.x * 128 + wave * 16;
    _Float16* aw = As[wave];
    for (int t = lane; t < 16 * DM; t += 32) {
        int r = t >> 7, c = t & 127;
        aw[r * LDA + c] = (_Float16)resp[(size_t)(mbase + r) * DM + c];
    }
    WAVE_LDS_FENCE();
    for (int nt = 0; nt < 8; ++nt) {
        float bv = bias[nt * 16 + (lane & 15)];
        v8f acc = {};
#pragma unroll
        for (int ks = 0; ks < 4; ++ks) {
            v16h bf = glbB(wt, 128, nt, ks, lane);
            v16h a  = ldsA(aw, LDA, 0, ks, lane);
            acc = wmma_f16(a, bf, acc);
        }
        int col = nt * 16 + (lane & 15);
        int rb  = (lane >> 4) * 8;
#pragma unroll
        for (int r = 0; r < 8; ++r) {
            size_t o = (size_t)(mbase + rb + r) * DM + col;
            out[o] = acc[r] + bv + xres[o];
        }
    }
}

// ---------- host ----------

extern "C" void kernel_launch(void* const* d_in, const int* in_sizes, int n_in,
                              void* d_out, int out_size, void* d_ws, size_t ws_size,
                              hipStream_t stream) {
    const float* features = (const float*)d_in[0];
    const float* xyz      = (const float*)d_in[1];
    const int*   knn_idx  = (const int*)d_in[2];
    const float* knn_xyz  = (const float*)d_in[3];
    const float* fc1_w = (const float*)d_in[4];
    const float* fc1_b = (const float*)d_in[5];
    const float* fc2_w = (const float*)d_in[6];
    const float* fc2_b = (const float*)d_in[7];
    const float* d1_w  = (const float*)d_in[8];
    const float* d1_b  = (const float*)d_in[9];
    const float* d2_w  = (const float*)d_in[10];
    const float* d2_b  = (const float*)d_in[11];
    const float* g1_w  = (const float*)d_in[12];
    const float* g1_b  = (const float*)d_in[13];
    const float* g2_w  = (const float*)d_in[14];
    const float* g2_b  = (const float*)d_in[15];

    float* out_res  = (float*)d_out;                       // (B,N,128)
    float* out_attn = out_res + (size_t)ROWS * DM;         // (B,N,24,128)
    char*  ws       = (char*)d_ws;
    float* xf       = (float*)(ws + WS_X);

    // allow >64KB dynamic LDS for the fused kernel (host-side, not a stream op)
    (void)hipFuncSetAttribute((const void*)pt_fused,
                              hipFuncAttributeMaxDynamicSharedMemorySize, SMEM_FUSED);

    prep_weights<<<320, 256, 0, stream>>>(fc1_w, d1_w, d2_w, g1_w, g2_w, fc2_w, ws);

    fc1_kernel<<<ROWS / 128, 256, 0, stream>>>(
        features, (const _Float16*)(ws + WS_WT_FC1), fc1_b, xf);

    pt_fused<<<ROWS / FUSED_WAVES, 32 * FUSED_WAVES, SMEM_FUSED, stream>>>(
        xyz, knn_idx, knn_xyz, xf,
        (const _Float16*)(ws + WS_WT_D1), d1_b,
        (const _Float16*)(ws + WS_WT_D2), d2_b,
        (const _Float16*)(ws + WS_WT_G1), g1_b,
        (const _Float16*)(ws + WS_WT_G2), g2_b,
        out_attn, out_res);

    fc2_kernel<<<ROWS / 128, 256, 0, stream>>>(
        out_res, (const _Float16*)(ws + WS_WT_FC2), fc2_b, xf, out_res);
}
